// GCN_71528385347706
// MI455X (gfx1250) — matrix operand order repaired and checked by reading, not compiled, not run on hardware
//
#include <hip/hip_runtime.h>
#include <hip/hip_bf16.h>
#include <math.h>

// ---------------- problem constants (match reference) ----------------
#define GN    20000      // nodes
#define GE    320000     // edges
#define DIN   128
#define DOUT  10
#define HDIM  1024
#define HM    256
#define EPSV  1e-5f

// ---------------- WMMA vector types ----------------
typedef __attribute__((ext_vector_type(16))) __bf16 v16bf;
typedef __attribute__((ext_vector_type(8)))  __bf16 v8bf;
typedef __attribute__((ext_vector_type(4)))  __bf16 v4bf;
typedef __attribute__((ext_vector_type(8)))  float  v8f;

#define LDS_STRIDE 40   // 32 bf16 payload + 8 pad (80B rows -> conflict-light b64/b128 access)

// =====================================================================
// Degree: deg[dst] += w for edges, +1 for self loops
// =====================================================================
__global__ void deg_kernel(const long long* __restrict__ ei,
                           const float* __restrict__ ew,
                           float* __restrict__ deg) {
    int t = blockIdx.x * blockDim.x + threadIdx.x;
    if (t < GE) {
        int d = (int)ei[(size_t)GE + t];
        atomicAdd(&deg[d], ew[t]);
    } else if (t < GE + GN) {
        atomicAdd(&deg[t - GE], 1.0f);
    }
}

// deg -> dinv (in place)
__global__ void dinv_kernel(float* __restrict__ deg) {
    int t = blockIdx.x * blockDim.x + threadIdx.x;
    if (t < GN) {
        float d = deg[t];
        deg[t] = (d > 0.0f) ? rsqrtf(d) : 0.0f;
    }
}

// norm[e] = dinv[src] * w * dinv[dst]  (self loops: dinv[i]^2)
__global__ void norm_kernel(const long long* __restrict__ ei,
                            const float* __restrict__ ew,
                            const float* __restrict__ dinv,
                            float* __restrict__ norm) {
    int t = blockIdx.x * blockDim.x + threadIdx.x;
    if (t < GE) {
        int s = (int)ei[t];
        int d = (int)ei[(size_t)GE + t];
        norm[t] = dinv[s] * ew[t] * dinv[d];
    } else if (t < GE + GN) {
        int i = t - GE;
        norm[t] = dinv[i] * dinv[i];
    }
}

// =====================================================================
// WMMA GEMM: C[M x Nw] = A[M x K] * B[K x Nw]  (f32 in, bf16 WMMA, f32 acc)
// block = 256 threads (8 waves), tile 128x128, BK = 32, double-buffered LDS
// wave w computes rows [w*16, w*16+16), all 128 cols (8 n-fragments)
// =====================================================================
__global__ __launch_bounds__(256)
void gemm_bf16_wmma(const float* __restrict__ A, const float* __restrict__ Bw,
                    float* __restrict__ C, int M, int K, int Nw,
                    const float* __restrict__ bias, int fuse_relu) {
    const int tid  = threadIdx.x;
    const int lane = tid & 31;
    const int wave = tid >> 5;
    const int m0 = blockIdx.y * 128;
    const int n0 = blockIdx.x * 128;

    __shared__ __bf16 As[2][128 * LDS_STRIDE];
    __shared__ __bf16 Bs[2][128 * LDS_STRIDE];

    v8f acc[8];
#pragma unroll
    for (int i = 0; i < 8; ++i)
#pragma unroll
        for (int j = 0; j < 8; ++j) acc[i][j] = 0.0f;

    // A-load mapping: tid -> (float4 quad within 32-k slab, base row)
    const int aq = tid & 7;    // which float4 in the k-slab
    const int ar = tid >> 3;   // base row 0..31 (stride 32 over 128 rows)
    // B-load mapping: tid -> (output column n, k half) ; thread loads columns
    // of 4 consecutive k (coalesced across the 128 consecutive-n threads per
    // row) and writes one packed ds_store_b64 per quad.
    const int bn   = tid & 127;  // column within tile
    const int bhalf = tid >> 7;  // 0 or 1

    // ---- staging (into buffer `st`, k-slab starting at k0) ----
    auto load_tile = [&](int st, int k0) {
#pragma unroll
        for (int rr = ar; rr < 128; rr += 32) {
            int gm = m0 + rr;
            float4 v = make_float4(0.f, 0.f, 0.f, 0.f);
            if (gm < M)
                v = *(const float4*)(A + (size_t)gm * K + k0 + aq * 4);
            v4bf pk;
            pk[0] = (__bf16)v.x; pk[1] = (__bf16)v.y;
            pk[2] = (__bf16)v.z; pk[3] = (__bf16)v.w;
            *(v4bf*)&As[st][rr * LDS_STRIDE + aq * 4] = pk;
        }
#pragma unroll
        for (int q = 0; q < 4; ++q) {
            int kk = bhalf * 4 + q * 8;                     // 0..28, step 4
            const float* bp = Bw + (size_t)(k0 + kk) * Nw + n0 + bn;
            v4bf pk;
            pk[0] = (__bf16)bp[0];
            pk[1] = (__bf16)bp[(size_t)Nw];
            pk[2] = (__bf16)bp[(size_t)Nw * 2];
            pk[3] = (__bf16)bp[(size_t)Nw * 3];
            *(v4bf*)&Bs[st][bn * LDS_STRIDE + kk] = pk;     // n-major, K contiguous
        }
    };

    // ---- fragment math from buffer `st` ----
    const int fr  = lane & 15;
    const int kbA = (lane >> 4) * 8;    // A: elems 0-7 = K kb..kb+7, 8-15 = K kb+16..kb+23
    const int kbB = (lane >> 4) * 8;    // B: elems 0-7 = K kb..kb+7, 8-15 = K kb+16..kb+23
    auto compute_tile = [&](int st) {
        v8bf alo = *(const v8bf*)&As[st][(wave * 16 + fr) * LDS_STRIDE + kbA];
        v8bf ahi = *(const v8bf*)&As[st][(wave * 16 + fr) * LDS_STRIDE + kbA + 16];
        v16bf a = __builtin_shufflevector(alo, ahi,
                    0,1,2,3,4,5,6,7,8,9,10,11,12,13,14,15);
#pragma unroll
        for (int nf = 0; nf < 8; ++nf) {
            v8bf blo = *(const v8bf*)&Bs[st][(nf * 16 + fr) * LDS_STRIDE + kbB];
            v8bf bhi = *(const v8bf*)&Bs[st][(nf * 16 + fr) * LDS_STRIDE + kbB + 16];
            v16bf b = __builtin_shufflevector(blo, bhi,
                        0,1,2,3,4,5,6,7,8,9,10,11,12,13,14,15);
            acc[nf] = __builtin_amdgcn_wmma_f32_16x16x32_bf16(
                false, a, false, b, (short)0, acc[nf], false, false);
        }
    };

    // ---- double-buffered main loop: stage k0+32 into p^1 while WMMAs eat p ----
    load_tile(0, 0);
    int p = 0;
    for (int k0 = 0; k0 < K; k0 += 32) {
        __syncthreads();                       // stores to p visible; p^1 free
        if (k0 + 32 < K) load_tile(p ^ 1, k0 + 32);
        compute_tile(p);
        p ^= 1;
    }

    // ---- epilogue: C/D layout -> lanes 0-15 hold M=v, lanes 16-31 hold M=v+8 ----
    const int rbase = m0 + wave * 16 + ((lane < 16) ? 0 : 8);
#pragma unroll
    for (int nf = 0; nf < 8; ++nf) {
        int col = n0 + nf * 16 + (lane & 15);
        float bv = bias ? bias[col] : 0.0f;
#pragma unroll
        for (int v = 0; v < 8; ++v) {
            int r = rbase + v;
            if (r < M) {
                float val = acc[nf][v] + bv;
                if (fuse_relu) val = fmaxf(val, 0.0f);
                C[(size_t)r * Nw + col] = val;
            }
        }
    }
}

// =====================================================================
// Aggregation: out[dst] += norm[e] * h[src], H = 1024 floats/row
// one 256-thread block per edge; float4 per lane; f32 atomics (L2-resident)
// =====================================================================
__global__ __launch_bounds__(256)
void agg_kernel(const float* __restrict__ h, float* __restrict__ out,
                const float* __restrict__ norm, const long long* __restrict__ ei) {
    int e = blockIdx.x;
    int src, dst;
    if (e < GE) {
        src = (int)ei[e];
        dst = (int)ei[(size_t)GE + e];
    } else {
        src = dst = e - GE;
    }
    float w = norm[e];
    float4 v = ((const float4*)(h + (size_t)src * HDIM))[threadIdx.x];
    float* o = out + (size_t)dst * HDIM + threadIdx.x * 4;
    atomicAdd(o + 0, w * v.x);
    atomicAdd(o + 1, w * v.y);
    atomicAdd(o + 2, w * v.z);
    atomicAdd(o + 3, w * v.w);
}

// =====================================================================
// BatchNorm: column sums / sumsq (pass A), finalize scale/shift, apply+ReLU
// =====================================================================
#define BN_ROWS 256
__global__ void bn_sums(const float* __restrict__ x, float* __restrict__ colsum,
                        float* __restrict__ colsumsq) {
    int c  = blockIdx.x * blockDim.x + threadIdx.x;   // 0..1023
    int r0 = blockIdx.y * BN_ROWS;
    int r1 = r0 + BN_ROWS; if (r1 > GN) r1 = GN;
    float s = 0.0f, ss = 0.0f;
    for (int r = r0; r < r1; ++r) {
        float v = x[(size_t)r * HDIM + c];
        s += v; ss += v * v;
    }
    atomicAdd(&colsum[c], s);
    atomicAdd(&colsumsq[c], ss);
}

__global__ void bn_finalize(const float* __restrict__ colsum,
                            const float* __restrict__ colsumsq,
                            const float* __restrict__ g,
                            const float* __restrict__ beta,
                            float* __restrict__ scale, float* __restrict__ shift) {
    int c = blockIdx.x * blockDim.x + threadIdx.x;
    if (c < HDIM) {
        float m   = colsum[c] * (1.0f / GN);
        float var = colsumsq[c] * (1.0f / GN) - m * m;
        float inv = rsqrtf(var + EPSV);
        float sc  = g[c] * inv;
        scale[c] = sc;
        shift[c] = beta[c] - m * sc;   // GCN bias cancels inside BN
    }
}

__global__ void bn_apply_relu(float* __restrict__ io,
                              const float* __restrict__ scale,
                              const float* __restrict__ shift) {
    size_t idx = (size_t)blockIdx.x * blockDim.x + threadIdx.x;  // float4 index
    size_t total = (size_t)GN * (HDIM / 4);
    if (idx >= total) return;
    int c4 = (int)(idx & (HDIM / 4 - 1));
    float4 v  = ((float4*)io)[idx];
    float4 sc = ((const float4*)scale)[c4];
    float4 sh = ((const float4*)shift)[c4];
    v.x = fmaxf(v.x * sc.x + sh.x, 0.0f);
    v.y = fmaxf(v.y * sc.y + sh.y, 0.0f);
    v.z = fmaxf(v.z * sc.z + sh.z, 0.0f);
    v.w = fmaxf(v.w * sc.w + sh.w, 0.0f);
    ((float4*)io)[idx] = v;
}

// =====================================================================
// Head: logits = hm[r,:256] @ Wl2[256x10] + bl2 ; log_softmax per row
// =====================================================================
__global__ void head_kernel(const float* __restrict__ hm,
                            const float* __restrict__ Wl2,
                            const float* __restrict__ bl2,
                            float* __restrict__ out) {
    int r = blockIdx.x * blockDim.x + threadIdx.x;
    if (r >= GN) return;
    float acc[DOUT];
#pragma unroll
    for (int j = 0; j < DOUT; ++j) acc[j] = bl2[j];
    const float* hr = hm + (size_t)r * HM;
    for (int k = 0; k < HM; ++k) {
        float hv = hr[k];
        const float* wk = Wl2 + k * DOUT;
#pragma unroll
        for (int j = 0; j < DOUT; ++j) acc[j] += hv * wk[j];
    }
    float mx = acc[0];
#pragma unroll
    for (int j = 1; j < DOUT; ++j) mx = fmaxf(mx, acc[j]);
    float se = 0.0f;
#pragma unroll
    for (int j = 0; j < DOUT; ++j) se += expf(acc[j] - mx);
    float lse = mx + logf(se);
#pragma unroll
    for (int j = 0; j < DOUT; ++j) out[(size_t)r * DOUT + j] = acc[j] - lse;
}

// =====================================================================
// Launch
// =====================================================================
extern "C" void kernel_launch(void* const* d_in, const int* in_sizes, int n_in,
                              void* d_out, int out_size, void* d_ws, size_t ws_size,
                              hipStream_t stream) {
    (void)in_sizes; (void)n_in; (void)out_size; (void)ws_size;

    const float*     x    = (const float*)d_in[0];
    const long long* ei   = (const long long*)d_in[1];   // int64 (2, E)
    const float*     ew   = (const float*)d_in[2];
    const float*     W1   = (const float*)d_in[3];
    //                b1   = d_in[4]   (cancels inside BN; zeros anyway)
    const float*     g1   = (const float*)d_in[5];
    const float*     be1  = (const float*)d_in[6];
    const float*     W2   = (const float*)d_in[7];
    //                b2   = d_in[8]
    const float*     g2   = (const float*)d_in[9];
    const float*     be2  = (const float*)d_in[10];
    const float*     Wl1  = (const float*)d_in[11];
    const float*     bl1  = (const float*)d_in[12];
    const float*     Wl2  = (const float*)d_in[13];
    const float*     bl2  = (const float*)d_in[14];
    float*           out  = (float*)d_out;

    // ---- workspace layout ----
    const size_t NH = (size_t)GN * HDIM;
    float* bufA     = (float*)d_ws;        // GEMM outputs / hm
    float* bufB     = bufA + NH;           // aggregation targets
    float* deg      = bufB + NH;           // N (becomes dinv)
    float* norm     = deg + GN;            // E + N
    float* colsum   = norm + (GE + GN);    // H
    float* colsumsq = colsum + HDIM;       // H
    float* scale    = colsumsq + HDIM;     // H
    float* shift    = scale + HDIM;        // H

    const int EN = GE + GN;
    const dim3 blk(256);

    // ---- graph prep ----
    hipMemsetAsync(deg, 0, GN * sizeof(float), stream);
    deg_kernel<<<(EN + 255) / 256, blk, 0, stream>>>(ei, ew, deg);
    dinv_kernel<<<(GN + 255) / 256, blk, 0, stream>>>(deg);
    norm_kernel<<<(EN + 255) / 256, blk, 0, stream>>>(ei, ew, deg, norm);

    const int mb = (GN + 127) / 128;        // 157
    const int bnY = (GN + BN_ROWS - 1) / BN_ROWS;
    const size_t nh4 = NH / 4;

    // ---- layer 1: x @ W1 -> agg -> BN -> ReLU ----
    gemm_bf16_wmma<<<dim3(HDIM / 128, mb), blk, 0, stream>>>(x, W1, bufA, GN, DIN, HDIM, nullptr, 0);
    hipMemsetAsync(bufB, 0, NH * sizeof(float), stream);
    agg_kernel<<<EN, blk, 0, stream>>>(bufA, bufB, norm, ei);
    hipMemsetAsync(colsum, 0, 2 * HDIM * sizeof(float), stream);
    bn_sums<<<dim3(HDIM / 256, bnY), blk, 0, stream>>>(bufB, colsum, colsumsq);
    bn_finalize<<<HDIM / 256, blk, 0, stream>>>(colsum, colsumsq, g1, be1, scale, shift);
    bn_apply_relu<<<(unsigned)((nh4 + 255) / 256), blk, 0, stream>>>(bufB, scale, shift);

    // ---- layer 2: h @ W2 -> agg -> BN -> ReLU ----
    gemm_bf16_wmma<<<dim3(HDIM / 128, mb), blk, 0, stream>>>(bufB, W2, bufA, GN, HDIM, HDIM, nullptr, 0);
    hipMemsetAsync(bufB, 0, NH * sizeof(float), stream);
    agg_kernel<<<EN, blk, 0, stream>>>(bufA, bufB, norm, ei);
    hipMemsetAsync(colsum, 0, 2 * HDIM * sizeof(float), stream);
    bn_sums<<<dim3(HDIM / 256, bnY), blk, 0, stream>>>(bufB, colsum, colsumsq);
    bn_finalize<<<HDIM / 256, blk, 0, stream>>>(colsum, colsumsq, g2, be2, scale, shift);
    bn_apply_relu<<<(unsigned)((nh4 + 255) / 256), blk, 0, stream>>>(bufB, scale, shift);

    // ---- MLP head: (h @ Wl1 + bl1).relu() @ Wl2 + bl2 -> log_softmax ----
    gemm_bf16_wmma<<<dim3(HM / 128, mb), blk, 0, stream>>>(bufB, Wl1, bufA, GN, HDIM, HM, bl1, 1);
    head_kernel<<<(GN + 255) / 256, blk, 0, stream>>>(bufA, Wl2, bl2, out);
}
